// Conditional_Entropy_Bottleneck_34050500722818
// MI455X (gfx1250) — compile-verified, matrix operand order checked by reading, not run on hardware
//
#include <hip/hip_runtime.h>
#include <hip/hip_bf16.h>

// ---------------------------------------------------------------------------
// Problem constants
// ---------------------------------------------------------------------------
#define BN 8192      // batch
#define DN 2048      // feature dim (K of the GEMM)
#define CN 1000      // classes
#define CP 1024      // classes padded to multiple of 256 for WMMA tiling

#define EPS32 1.1920928955078125e-07f
#define GAMMA 3.7200759760208356e-44f   // exp(-100), f32 denormal

typedef __bf16 bf16_t;
typedef __attribute__((ext_vector_type(16))) bf16_t v16bf;
typedef __attribute__((ext_vector_type(8)))  float  v8f;

// round-to-nearest-even f32 -> bf16 bit pattern (no __bf16 arithmetic needed)
__device__ __forceinline__ unsigned short f32_to_bf16(float x) {
    unsigned int u = __float_as_uint(x);
    u += 0x7FFFu + ((u >> 16) & 1u);
    return (unsigned short)(u >> 16);
}

// ---------------------------------------------------------------------------
// Kernel 1: z = feature + eps (store bf16), per-row reductions for h2-h1
//   rowterm[b] = GAMMA * 0.5 * ( sum (z - y_emb)^2 - sum eps^2 )
// ---------------------------------------------------------------------------
__global__ __launch_bounds__(256)
void prep_z_kernel(const float* __restrict__ feature,
                   const float* __restrict__ eps,
                   const float* __restrict__ y_w,   // [D, C]
                   const float* __restrict__ y_b,   // [D]
                   const int*   __restrict__ target,
                   unsigned short* __restrict__ zb, // [B, D] bf16 bits
                   float* __restrict__ rowterm) {
    const int b   = blockIdx.x;
    const int tid = threadIdx.x;
    const int t   = target[b];
    const float* frow = feature + (size_t)b * DN;
    const float* erow = eps     + (size_t)b * DN;
    unsigned short* zrow = zb + (size_t)b * DN;

    float se = 0.0f, sd = 0.0f;
#pragma unroll
    for (int i = 0; i < DN / 256; ++i) {
        const int d = tid + i * 256;
        const float e = erow[d];
        const float z = frow[d] + e;
        zrow[d] = f32_to_bf16(z);
        const float ye = y_w[(size_t)d * CN + t] + y_b[d];
        const float dz = z - ye;
        se += e * e;
        sd += dz * dz;
    }

    __shared__ float r1[256];
    __shared__ float r2[256];
    r1[tid] = se;
    r2[tid] = sd;
    __syncthreads();
    for (int off = 128; off > 0; off >>= 1) {
        if (tid < off) { r1[tid] += r1[tid + off]; r2[tid] += r2[tid + off]; }
        __syncthreads();
    }
    if (tid == 0) rowterm[b] = GAMMA * 0.5f * (r2[0] - r1[0]);
}

// ---------------------------------------------------------------------------
// Kernel 2: cls_w [C, D] f32 -> [CP, D] bf16, rows >= CN zero padded
// ---------------------------------------------------------------------------
__global__ __launch_bounds__(256)
void prep_w_kernel(const float* __restrict__ cls_w,
                   unsigned short* __restrict__ wb) {
    const size_t i = (size_t)blockIdx.x * 256 + threadIdx.x;  // over CP*DN
    const int c = (int)(i >> 11);          // / DN
    const int d = (int)(i & (DN - 1));
    const float v = (c < CN) ? cls_w[(size_t)c * DN + d] : 0.0f;
    wb[i] = f32_to_bf16(v);
}

// ---------------------------------------------------------------------------
// Kernel 3: WMMA GEMM  praw[B, CP] = z_bf16 [B, DN] @ w_bf16 [CP, DN]^T
//   grid = (64, 4), block = 256 threads = 8 waves, wave tile = 64x64
//   A/B fragment layout per CDNA5 ISA 7.12.2 (16-bit, 16x32 / 32x16)
// ---------------------------------------------------------------------------
__global__ __launch_bounds__(256)
void gemm_kernel(const unsigned short* __restrict__ zb,
                 const unsigned short* __restrict__ wb,
                 float* __restrict__ praw) {
    const int wave = threadIdx.x >> 5;   // 0..7
    const int lane = threadIdx.x & 31;
    const int half = lane >> 4;          // 0: K lo 8 / 1: K hi 8 of each 16
    const int l15  = lane & 15;

    const int m0 = blockIdx.x * 128 + (wave >> 2) * 64;  // output rows
    const int n0 = blockIdx.y * 256 + (wave & 3) * 64;   // output cols

    // per-lane row base pointers (A: rows of z, B: rows of cls_w == cols of B)
    const unsigned short* arow[4];
    const unsigned short* brow[4];
#pragma unroll
    for (int i = 0; i < 4; ++i) {
        arow[i] = zb + (size_t)(m0 + i * 16 + l15) * DN;
        brow[i] = wb + (size_t)(n0 + i * 16 + l15) * DN;
    }

    v8f acc[4][4];
#pragma unroll
    for (int i = 0; i < 4; ++i)
#pragma unroll
        for (int j = 0; j < 4; ++j)
            acc[i][j] = (v8f)0.0f;

    union Frag { v16bf v; uint4 q[2]; };

    for (int kc = 0; kc < DN; kc += 32) {
        const int k0 = kc + half * 8;   // element offset, 16B aligned

        Frag a[4], bfr[4];
#pragma unroll
        for (int i = 0; i < 4; ++i) {
            a[i].q[0] = *(const uint4*)(arow[i] + k0);
            a[i].q[1] = *(const uint4*)(arow[i] + k0 + 16);
        }
#pragma unroll
        for (int j = 0; j < 4; ++j) {
            bfr[j].q[0] = *(const uint4*)(brow[j] + k0);
            bfr[j].q[1] = *(const uint4*)(brow[j] + k0 + 16);
        }

        // hint next K-chunk into cache while the matrix pipe works
        if (kc + 32 < DN) {
            __builtin_prefetch(arow[0] + k0 + 32, 0, 3);
            __builtin_prefetch(brow[0] + k0 + 32, 0, 3);
        }

#pragma unroll
        for (int i = 0; i < 4; ++i)
#pragma unroll
            for (int j = 0; j < 4; ++j)
                acc[i][j] = __builtin_amdgcn_wmma_f32_16x16x32_bf16(
                    /*neg_a=*/false, a[i].v,
                    /*neg_b=*/false, bfr[j].v,
                    /*c_mod=*/(short)0, acc[i][j],
                    /*reuse_a=*/false, /*reuse_b=*/false);
    }

    // C/D layout: VGPR v, lanes 0-15 -> (M = v,     N = lane)
    //                     lanes 16-31 -> (M = v + 8, N = lane - 16)
#pragma unroll
    for (int i = 0; i < 4; ++i) {
#pragma unroll
        for (int j = 0; j < 4; ++j) {
            const int rbase = m0 + i * 16 + 8 * half;
            const int col   = n0 + j * 16 + l15;
#pragma unroll
            for (int v = 0; v < 8; ++v)
                praw[(size_t)(rbase + v) * CP + col] = acc[i][j][v];
        }
    }
}

// ---------------------------------------------------------------------------
// Kernel 4: per-row normalize + clamp + log, write logits, fold in h3
// ---------------------------------------------------------------------------
__global__ __launch_bounds__(256)
void finalize_kernel(const float* __restrict__ praw,
                     const float* __restrict__ cls_b,
                     const int*   __restrict__ target,
                     float* __restrict__ rowterm,
                     float* __restrict__ out_logits) {
    const int b   = blockIdx.x;
    const int tid = threadIdx.x;
    const int t   = target[b];
    const float* row = praw + (size_t)b * CP;

    float s = 0.0f;
    for (int c = tid; c < CN; c += 256) s += row[c] + cls_b[c];

    __shared__ float red[256];
    __shared__ float h3s;
    red[tid] = s;
    __syncthreads();
    for (int off = 128; off > 0; off >>= 1) {
        if (tid < off) red[tid] += red[tid + off];
        __syncthreads();
    }
    const float inv = 1.0f / red[0];

    for (int c = tid; c < CN; c += 256) {
        float p = (row[c] + cls_b[c]) * inv;
        p = fminf(fmaxf(p, EPS32), 1.0f - EPS32);
        const float lg = __logf(p);
        out_logits[(size_t)b * CN + c] = lg;
        if (c == t) h3s = -lg;
    }
    __syncthreads();
    if (tid == 0) rowterm[b] += h3s;
}

// ---------------------------------------------------------------------------
// Kernel 5: deterministic mean over rowterm -> loss
// ---------------------------------------------------------------------------
__global__ __launch_bounds__(256)
void loss_kernel(const float* __restrict__ rowterm, float* __restrict__ out_loss) {
    __shared__ float red[256];
    float s = 0.0f;
    for (int i = threadIdx.x; i < BN; i += 256) s += rowterm[i];
    red[threadIdx.x] = s;
    __syncthreads();
    for (int off = 128; off > 0; off >>= 1) {
        if (threadIdx.x < off) red[threadIdx.x] += red[threadIdx.x + off];
        __syncthreads();
    }
    if (threadIdx.x == 0) *out_loss = red[0] * (1.0f / (float)BN);
}

// ---------------------------------------------------------------------------
// Launch
// ---------------------------------------------------------------------------
extern "C" void kernel_launch(void* const* d_in, const int* in_sizes, int n_in,
                              void* d_out, int out_size, void* d_ws, size_t ws_size,
                              hipStream_t stream) {
    (void)in_sizes; (void)n_in; (void)out_size; (void)ws_size;

    const float* feature = (const float*)d_in[0];  // [B, D]
    const float* y_w     = (const float*)d_in[1];  // [D, C]
    const float* y_b     = (const float*)d_in[2];  // [D]
    const float* cls_w   = (const float*)d_in[3];  // [C, D]
    const float* cls_b   = (const float*)d_in[4];  // [C]
    const float* eps     = (const float*)d_in[5];  // [B, D]
    const int*   target  = (const int*)  d_in[6];  // [B]

    float* out_logits = (float*)d_out;                       // [B, C]
    float* out_loss   = (float*)d_out + (size_t)BN * CN;     // scalar

    // workspace carve-up
    char* ws = (char*)d_ws;
    unsigned short* zb   = (unsigned short*)ws;                              // 32 MB
    unsigned short* wb   = (unsigned short*)(ws + (size_t)BN * DN * 2);      //  4 MB
    float*          praw = (float*)(ws + (size_t)BN * DN * 2
                                       + (size_t)CP * DN * 2);               // 32 MB
    float*          rowterm = (float*)(ws + (size_t)BN * DN * 2
                                          + (size_t)CP * DN * 2
                                          + (size_t)BN * CP * 4);            // 32 KB

    prep_z_kernel<<<BN, 256, 0, stream>>>(feature, eps, y_w, y_b, target, zb, rowterm);
    prep_w_kernel<<<(CP * DN) / 256, 256, 0, stream>>>(cls_w, wb);
    gemm_kernel<<<dim3(BN / 128, CP / 256), 256, 0, stream>>>(zb, wb, praw);
    finalize_kernel<<<BN, 256, 0, stream>>>(praw, cls_b, target, rowterm, out_logits);
    loss_kernel<<<1, 256, 0, stream>>>(rowterm, out_loss);
}
